// VGG9_Direct_58067957842464
// MI455X (gfx1250) — matrix operand order repaired and checked by reading, not compiled, not run on hardware
//
#include <hip/hip_runtime.h>

typedef __attribute__((ext_vector_type(16))) _Float16 v16h;
typedef __attribute__((ext_vector_type(8)))  float    v8f;
typedef _Float16 h16;

#define BATCH 128

union FragH { v16h v; unsigned int u[8]; };
union Out8  { uint4 q; h16 h[8]; };

// ---------------- elementwise helpers ----------------

__global__ void zero_f32_kernel(float* p, int n) {
  int i = blockIdx.x * 256 + threadIdx.x;
  if (i < n) p[i] = 0.0f;
}

// conv weights: OIHW f32 -> [Cout][tap*Cin+ci] f16 (tap-major K)
__global__ void reorder_conv_w_kernel(const float* __restrict__ src, h16* __restrict__ dst,
                                      int Cout, int Cin) {
  int idx = blockIdx.x * 256 + threadIdx.x;
  int total = Cout * Cin * 9;
  if (idx >= total) return;
  int tap = idx % 9;
  int ci  = (idx / 9) % Cin;
  int m   = idx / (9 * Cin);
  dst[(size_t)m * (9 * Cin) + tap * Cin + ci] = (h16)src[idx];
}

// fc1 weights: [1024][c*16+p] f32 -> [1024][p*256+c] f16 (match NHWC flatten)
__global__ void reorder_fc1_w_kernel(const float* __restrict__ src, h16* __restrict__ dst) {
  int idx = blockIdx.x * 256 + threadIdx.x;   // 1024*4096 exact
  int k = idx & 4095;
  int m = idx >> 12;
  int c = k >> 4;
  int p = k & 15;
  dst[((size_t)m << 12) + p * 256 + c] = (h16)src[idx];
}

// conv1: 3->64, 32x32, pad 1, fp32, NCHW input -> NHWC output (static input)
__global__ void conv1_static_kernel(const float* __restrict__ inp,
                                    const float* __restrict__ w1,
                                    float* __restrict__ out) {
  int idx = blockIdx.x * 256 + threadIdx.x;  // 128*32*32*64 = 8388608 exact
  int co = idx & 63;
  int x  = (idx >> 6) & 31;
  int y  = (idx >> 11) & 31;
  int b  = idx >> 16;
  float acc = 0.0f;
  for (int ci = 0; ci < 3; ++ci)
    for (int r = 0; r < 3; ++r) {
      int iy = y + r - 1;
      if ((unsigned)iy >= 32u) continue;
      for (int s = 0; s < 3; ++s) {
        int ix = x + s - 1;
        if ((unsigned)ix >= 32u) continue;
        acc += inp[((b * 3 + ci) * 32 + iy) * 32 + ix] *
               w1[((co * 3 + ci) * 3 + r) * 3 + s];
      }
    }
  out[idx] = acc;  // NHWC
}

// LIF for conv1 membrane (layout-agnostic elementwise)
__global__ void lif_conv1_kernel(const float* __restrict__ stat,
                                 float* __restrict__ mem0,
                                 h16* __restrict__ act) {
  int i = blockIdx.x * 256 + threadIdx.x;    // 8388608 exact
  float m = 0.05f * mem0[i] + 0.95f * stat[i];
  bool s = m > 1.0f;
  act[i]  = s ? (h16)1.0f : (h16)0.0f;
  mem0[i] = s ? (m - 1.0f) : m;
}

// 2x2 average pool, NHWC f16
__global__ void avgpool2_nhwc_kernel(const h16* __restrict__ in, h16* __restrict__ out,
                                     int C, int Ho, int Wo) {
  int i = blockIdx.x * 256 + threadIdx.x;
  int total = BATCH * Ho * Wo * C;
  if (i >= total) return;
  int c = i % C;
  int x = (i / C) % Wo;
  int y = (i / (C * Wo)) % Ho;
  int b = i / (C * Wo * Ho);
  int W = Wo * 2, H = Ho * 2;
  const h16* p = in + (((size_t)(b * H + 2 * y) * W) + 2 * x) * C + c;
  size_t rowC = (size_t)W * C;
  float v = (float)p[0] + (float)p[C] + (float)p[rowC] + (float)p[rowC + C];
  out[i] = (h16)(0.25f * v);
}

// ---------------- implicit-GEMM conv with WMMA (NHWC, tap-major K) ----------
// D[m][n] = sum_k W[m][k] * Patch[k][n]; k = tap*Cin+ci, n = (b,y,x)
// Block: 256 thr = 8 waves; block tile 64(M) x 128(N); wave tile 32x32.
// K-step 64: 8 wmma per staging round (2 barriers / 8 wmma).
__global__ __launch_bounds__(256) void conv_gemm_wmma_kernel(
    const h16* __restrict__ in, const h16* __restrict__ w, h16* __restrict__ out,
    int Cin, int Cout, int H, int W) {
  __shared__ __align__(16) h16 As[64 * 64];    // [m][k]  8KB
  __shared__ __align__(16) h16 Bs[128 * 64];   // [n][k] 16KB
  const int K  = Cin * 9;
  const int HW = H * W;
  const int n0 = blockIdx.x * 128;
  const int m0 = blockIdx.y * 64;
  const int t    = threadIdx.x;
  const int lane = t & 31, wid = t >> 5;
  const int g = lane >> 4, l16 = lane & 15;
  const int wm = wid & 1, wn = wid >> 1;       // 2 (M) x 4 (N) waves

  // A loader: row am (0..63), 16 halves at ak
  const int am = t >> 2, ak = (t & 3) * 16;
  // B loader: col bn (0..127), 32 halves at bk
  const int bn = t >> 1, bk = (t & 1) * 32;
  const int ngth = n0 + bn;
  const int bb   = ngth / HW;
  const int remt = ngth - bb * HW;
  const int yy   = remt / W;
  const int xx   = remt - yy * W;

  v8f acc00 = {}, acc01 = {}, acc10 = {}, acc11 = {};
  const uint4 zero4 = {0u, 0u, 0u, 0u};

  const int ar0 = (wm * 32 + l16) * 64;
  const int ar1 = (wm * 32 + 16 + l16) * 64;
  const int br0 = (wn * 32 + l16) * 64 + 16 * g;
  const int br1 = (wn * 32 + 16 + l16) * 64 + 16 * g;

  for (int tap = 0; tap < 9; ++tap) {
    const int dy = tap / 3 - 1;
    const int dx = tap - (tap / 3) * 3 - 1;
    const int iy = yy + dy, ix = xx + dx;
    const bool inb = ((unsigned)iy < (unsigned)H) && ((unsigned)ix < (unsigned)W);
    const h16* src  = in + ((size_t)(bb * H + iy) * W + ix) * Cin + bk;
    const h16* wrow = w + (size_t)(m0 + am) * K + tap * Cin + ak;

    for (int cc = 0; cc < Cin; cc += 64) {
      // A tile: 64x64 halves, two uint4 per thread
      *(uint4*)(&As[am * 64 + ak])     = *(const uint4*)(wrow + cc);
      *(uint4*)(&As[am * 64 + ak + 8]) = *(const uint4*)(wrow + cc + 8);
      // B tile: 128x64 halves, four uint4 per thread (contiguous channels)
      uint4 v0 = zero4, v1 = zero4, v2 = zero4, v3 = zero4;
      if (inb) {
        v0 = *(const uint4*)(src + cc);
        v1 = *(const uint4*)(src + cc + 8);
        v2 = *(const uint4*)(src + cc + 16);
        v3 = *(const uint4*)(src + cc + 24);
      }
      *(uint4*)(&Bs[bn * 64 + bk])      = v0;
      *(uint4*)(&Bs[bn * 64 + bk + 8])  = v1;
      *(uint4*)(&Bs[bn * 64 + bk + 16]) = v2;
      *(uint4*)(&Bs[bn * 64 + bk + 24]) = v3;
      __syncthreads();

      #pragma unroll
      for (int kh = 0; kh < 2; ++kh) {      // two 32-wide K halves
        const int ko = kh * 32;
        FragH a0, a1, b0, b1;
        #pragma unroll
        for (int r = 0; r < 8; ++r) {
          int kloc = 16 * g + 2 * r;  // contiguous k = 16g + s (matches B)
          a0.u[r] = *(const unsigned int*)(&As[ar0 + ko + kloc]);
          a1.u[r] = *(const unsigned int*)(&As[ar1 + ko + kloc]);
        }
        #pragma unroll
        for (int r = 0; r < 8; ++r) {                            // contiguous k = 16g + s
          b0.u[r] = *(const unsigned int*)(&Bs[br0 + ko + 2 * r]);
          b1.u[r] = *(const unsigned int*)(&Bs[br1 + ko + 2 * r]);
        }
        acc00 = __builtin_amdgcn_wmma_f32_16x16x32_f16(false, a0.v, false, b0.v, (short)0, acc00, false, false);
        acc01 = __builtin_amdgcn_wmma_f32_16x16x32_f16(false, a0.v, false, b1.v, (short)0, acc01, false, false);
        acc10 = __builtin_amdgcn_wmma_f32_16x16x32_f16(false, a1.v, false, b0.v, (short)0, acc10, false, false);
        acc11 = __builtin_amdgcn_wmma_f32_16x16x32_f16(false, a1.v, false, b1.v, (short)0, acc11, false, false);
      }
      __syncthreads();
    }
  }

  // epilogue: spike threshold, one 16B NHWC store per subtile
  v8f accs[2][2] = {{acc00, acc01}, {acc10, acc11}};
  #pragma unroll
  for (int ms = 0; ms < 2; ++ms) {
    #pragma unroll
    for (int ns = 0; ns < 2; ++ns) {
      int n  = n0 + wn * 32 + ns * 16 + l16;
      int b  = n / HW;
      int rm = n - b * HW;
      int mb = m0 + wm * 32 + ms * 16 + 8 * g;   // 8 consecutive channels
      Out8 o;
      #pragma unroll
      for (int j = 0; j < 8; ++j)
        o.h[j] = (accs[ms][ns][j] > 1.0f) ? (h16)1.0f : (h16)0.0f;
      *(uint4*)(&out[((size_t)b * HW + rm) * Cout + mb]) = o.q;
    }
  }
}

// ---------------- FC1 GEMM with WMMA + fused LIF ----------------
// z[n][m] = sum_k act[n][k] * Wfc[m][k]; M=1024, N=128, K=4096 (weights pre-permuted)
__global__ __launch_bounds__(256) void fc1_gemm_wmma_kernel(
    const h16* __restrict__ act, const h16* __restrict__ w,
    float* __restrict__ mem, h16* __restrict__ outact) {
  __shared__ __align__(16) h16 As[64 * 64];
  __shared__ __align__(16) h16 Bs[128 * 64];
  const int K  = 4096;
  const int n0 = 0;                  // N=128 == block tile
  const int m0 = blockIdx.y * 64;
  const int t    = threadIdx.x;
  const int lane = t & 31, wid = t >> 5;
  const int g = lane >> 4, l16 = lane & 15;
  const int wm = wid & 1, wn = wid >> 1;

  const int am = t >> 2, ak = (t & 3) * 16;
  const int bn = t >> 1, bk = (t & 1) * 32;

  v8f acc00 = {}, acc01 = {}, acc10 = {}, acc11 = {};

  const int ar0 = (wm * 32 + l16) * 64;
  const int ar1 = (wm * 32 + 16 + l16) * 64;
  const int br0 = (wn * 32 + l16) * 64 + 16 * g;
  const int br1 = (wn * 32 + 16 + l16) * 64 + 16 * g;

  for (int kk = 0; kk < K; kk += 64) {
    *(uint4*)(&As[am * 64 + ak])     = *(const uint4*)(&w[(size_t)(m0 + am) * K + kk + ak]);
    *(uint4*)(&As[am * 64 + ak + 8]) = *(const uint4*)(&w[(size_t)(m0 + am) * K + kk + ak + 8]);
    const h16* asrc = act + (size_t)(n0 + bn) * K + kk + bk;
    *(uint4*)(&Bs[bn * 64 + bk])      = *(const uint4*)(asrc);
    *(uint4*)(&Bs[bn * 64 + bk + 8])  = *(const uint4*)(asrc + 8);
    *(uint4*)(&Bs[bn * 64 + bk + 16]) = *(const uint4*)(asrc + 16);
    *(uint4*)(&Bs[bn * 64 + bk + 24]) = *(const uint4*)(asrc + 24);
    __syncthreads();

    #pragma unroll
    for (int kh = 0; kh < 2; ++kh) {
      const int ko = kh * 32;
      FragH a0, a1, b0, b1;
      #pragma unroll
      for (int r = 0; r < 8; ++r) {
        int kloc = 16 * g + 2 * r;
        a0.u[r] = *(const unsigned int*)(&As[ar0 + ko + kloc]);
        a1.u[r] = *(const unsigned int*)(&As[ar1 + ko + kloc]);
      }
      #pragma unroll
      for (int r = 0; r < 8; ++r) {
        b0.u[r] = *(const unsigned int*)(&Bs[br0 + ko + 2 * r]);
        b1.u[r] = *(const unsigned int*)(&Bs[br1 + ko + 2 * r]);
      }
      acc00 = __builtin_amdgcn_wmma_f32_16x16x32_f16(false, a0.v, false, b0.v, (short)0, acc00, false, false);
      acc01 = __builtin_amdgcn_wmma_f32_16x16x32_f16(false, a0.v, false, b1.v, (short)0, acc01, false, false);
      acc10 = __builtin_amdgcn_wmma_f32_16x16x32_f16(false, a1.v, false, b0.v, (short)0, acc10, false, false);
      acc11 = __builtin_amdgcn_wmma_f32_16x16x32_f16(false, a1.v, false, b1.v, (short)0, acc11, false, false);
    }
    __syncthreads();
  }

  // fused LIF on mem_fc1 [n][1024]
  v8f accs[2][2] = {{acc00, acc01}, {acc10, acc11}};
  #pragma unroll
  for (int ms = 0; ms < 2; ++ms) {
    #pragma unroll
    for (int ns = 0; ns < 2; ++ns) {
      int n  = n0 + wn * 32 + ns * 16 + l16;
      int mb = m0 + wm * 32 + ms * 16 + 8 * g;
      #pragma unroll
      for (int j = 0; j < 8; ++j) {
        int o = n * 1024 + mb + j;
        float mm = 0.05f * mem[o] + 0.95f * accs[ms][ns][j];
        bool s = mm > 1.0f;
        outact[o] = s ? (h16)1.0f : (h16)0.0f;
        mem[o] = s ? (mm - 1.0f) : mm;
      }
    }
  }
}

// FC2: mem_fc2[n][c] += sum_k act[n][k] * w[c][k]   (128x10, tiny)
__global__ void fc2_acc_kernel(const h16* __restrict__ act,
                               const float* __restrict__ w,
                               float* __restrict__ mem2) {
  int i = blockIdx.x * 256 + threadIdx.x;
  if (i >= BATCH * 10) return;
  int n = i / 10, c = i % 10;
  const h16* a = act + (size_t)n * 1024;
  const float* ww = w + (size_t)c * 1024;
  float acc = 0.0f;
  for (int k = 0; k < 1024; ++k) acc += (float)a[k] * ww[k];
  mem2[i] += acc;
}

__global__ void finalize_kernel(const float* __restrict__ mem2, float* __restrict__ out) {
  int i = blockIdx.x * 256 + threadIdx.x;
  if (i < BATCH * 10) out[i] = mem2[i] * 0.1f;
}

// ---------------- host launch ----------------

extern "C" void kernel_launch(void* const* d_in, const int* in_sizes, int n_in,
                              void* d_out, int out_size, void* d_ws, size_t ws_size,
                              hipStream_t stream) {
  (void)in_sizes; (void)n_in; (void)out_size; (void)ws_size;
  const float* inp  = (const float*)d_in[0];
  const float* w1   = (const float*)d_in[1];
  const float* wc[6] = { (const float*)d_in[2], (const float*)d_in[3],
                         (const float*)d_in[4], (const float*)d_in[5],
                         (const float*)d_in[6], (const float*)d_in[7] };
  const float* fc1w = (const float*)d_in[8];
  const float* fc2w = (const float*)d_in[9];

  char* ws = (char*)d_ws;
  size_t cur = 0;
  auto alloc = [&](size_t bytes) -> void* {
    void* p = ws + cur;
    cur += (bytes + 255) & ~(size_t)255;
    return p;
  };

  const int NSTAT = BATCH * 64 * 32 * 32;        // 8388608
  float* stat   = (float*)alloc((size_t)NSTAT * 4);
  float* mem0   = (float*)alloc((size_t)NSTAT * 4);
  h16*   actA   = (h16*)  alloc((size_t)NSTAT * 2);
  h16*   actB   = (h16*)  alloc((size_t)NSTAT * 2);
  const int cinA[6]  = { 64, 64, 128, 128, 256, 256 };
  const int coutA[6] = { 64, 128, 128, 256, 256, 256 };
  h16* wch[6];
  for (int i = 0; i < 6; ++i) wch[i] = (h16*)alloc((size_t)coutA[i] * cinA[i] * 9 * 2);
  h16*   fc1h   = (h16*)  alloc((size_t)1024 * 4096 * 2);
  float* memfc1 = (float*)alloc((size_t)BATCH * 1024 * 4);
  float* memfc2 = (float*)alloc((size_t)BATCH * 10 * 4);
  h16*   fc1act = (h16*)  alloc((size_t)BATCH * 1024 * 2);

  // zero LIF state
  zero_f32_kernel<<<(NSTAT + 255) / 256, 256, 0, stream>>>(mem0, NSTAT);
  zero_f32_kernel<<<(BATCH * 1024 + 255) / 256, 256, 0, stream>>>(memfc1, BATCH * 1024);
  zero_f32_kernel<<<(BATCH * 10 + 255) / 256, 256, 0, stream>>>(memfc2, BATCH * 10);

  // reorder + convert weights to f16 (tap-major / NHWC-flatten-major)
  for (int i = 0; i < 6; ++i) {
    int tot = coutA[i] * cinA[i] * 9;
    reorder_conv_w_kernel<<<(tot + 255) / 256, 256, 0, stream>>>(wc[i], wch[i], coutA[i], cinA[i]);
  }
  reorder_fc1_w_kernel<<<(1024 * 4096) / 256, 256, 0, stream>>>(fc1w, fc1h);

  // static input (conv1, NHWC) once
  conv1_static_kernel<<<NSTAT / 256, 256, 0, stream>>>(inp, w1, stat);

  for (int step = 0; step < 10; ++step) {
    lif_conv1_kernel<<<NSTAT / 256, 256, 0, stream>>>(stat, mem0, actA);
    // conv2: 64->64 @32x32
    conv_gemm_wmma_kernel<<<dim3(BATCH*32*32/128, 64/64), 256, 0, stream>>>(actA, wch[0], actB, 64, 64, 32, 32);
    avgpool2_nhwc_kernel<<<(BATCH*16*16*64 + 255) / 256, 256, 0, stream>>>(actB, actA, 64, 16, 16);
    // conv3: 64->128 @16x16
    conv_gemm_wmma_kernel<<<dim3(BATCH*16*16/128, 128/64), 256, 0, stream>>>(actA, wch[1], actB, 64, 128, 16, 16);
    // conv4: 128->128 @16x16
    conv_gemm_wmma_kernel<<<dim3(BATCH*16*16/128, 128/64), 256, 0, stream>>>(actB, wch[2], actA, 128, 128, 16, 16);
    avgpool2_nhwc_kernel<<<(BATCH*8*8*128 + 255) / 256, 256, 0, stream>>>(actA, actB, 128, 8, 8);
    // conv5: 128->256 @8x8
    conv_gemm_wmma_kernel<<<dim3(BATCH*8*8/128, 256/64), 256, 0, stream>>>(actB, wch[3], actA, 128, 256, 8, 8);
    // conv6: 256->256 @8x8
    conv_gemm_wmma_kernel<<<dim3(BATCH*8*8/128, 256/64), 256, 0, stream>>>(actA, wch[4], actB, 256, 256, 8, 8);
    // conv7: 256->256 @8x8
    conv_gemm_wmma_kernel<<<dim3(BATCH*8*8/128, 256/64), 256, 0, stream>>>(actB, wch[5], actA, 256, 256, 8, 8);
    avgpool2_nhwc_kernel<<<(BATCH*4*4*256 + 255) / 256, 256, 0, stream>>>(actA, actB, 256, 4, 4);
    // fc1 (WMMA) + fused LIF; pool3 NHWC flat matches permuted weights
    fc1_gemm_wmma_kernel<<<dim3(1, 1024/64), 256, 0, stream>>>(actB, fc1h, memfc1, fc1act);
    fc2_acc_kernel<<<(BATCH*10 + 255) / 256, 256, 0, stream>>>(fc1act, fc2w, memfc2);
  }

  finalize_kernel<<<(BATCH*10 + 255) / 256, 256, 0, stream>>>(memfc2, (float*)d_out);
}
